// FeatureAlign_V2_41300405518790
// MI455X (gfx1250) — matrix operand order repaired and checked by reading, not compiled, not run on hardware
//
#include <hip/hip_runtime.h>
#include <hip/hip_bf16.h>
#include <math.h>

// ---------------------------------------------------------------------------
// FeatureAlign_V2 for MI455X (gfx1250, wave32, WMMA).
// All convs run as v_wmma_f32_16x16x32_f16 GEMMs (f16 operands, f32 accum).
// Activations for GEMMs are staged pixel-major [p][K] so every B fragment is
// 16 contiguous f16 (2x global_load_b128); weights row-major so every A
// fragment is 2x b128. gemm1x1: 4 N-subtiles/wave; conv3x3: 6 subtiles (one
// full image row) per wave for A reuse. Deform-conv gathers stay L2-resident.
// ---------------------------------------------------------------------------

typedef __attribute__((ext_vector_type(16))) _Float16 v16h;
typedef __attribute__((ext_vector_type(8)))  _Float16 v8h;
typedef __attribute__((ext_vector_type(8)))  float    v8f;

#define HW   9216     // 96*96
#define WID  96
#define NB   2
#define NC   256
#define EPSV 1e-5f

__device__ __forceinline__ int imin(int a, int b) { return a < b ? a : b; }
__device__ __forceinline__ int imax(int a, int b) { return a > b ? a : b; }
__device__ __forceinline__ float sigm(float x) { return 1.f / (1.f + expf(-x)); }

// A fragment (16x32 f16): elem e = K half*16+e -> 16 contiguous f16 per lane
__device__ __forceinline__ v16h load_a_frag(const _Float16* __restrict__ rowk0, int half) {
  v8h lo = *(const v8h*)(rowk0 + half * 16);
  v8h hi = *(const v8h*)(rowk0 + half * 16 + 8);
  v16h a;
#pragma unroll
  for (int e = 0; e < 8; ++e) { a[e] = lo[e]; a[8 + e] = hi[e]; }
  return a;
}

// B fragment (32x16 f16): elem e = K half*16+e  -> 16 contiguous f16 per lane
__device__ __forceinline__ v16h load_b_frag(const _Float16* __restrict__ rowk0, int half) {
  v8h lo = *(const v8h*)(rowk0 + half * 16);
  v8h hi = *(const v8h*)(rowk0 + half * 16 + 8);
  v16h b;
#pragma unroll
  for (int e = 0; e < 8; ++e) { b[e] = lo[e]; b[8 + e] = hi[e]; }
  return b;
}

__device__ __forceinline__ v16h zero16() { v16h z = {}; return z; }

// ------------------------- elementwise / small kernels ----------------------

__global__ __launch_bounds__(256) void cvt_f16_kernel(const float* __restrict__ in,
                                                      _Float16* __restrict__ out, int n) {
  for (int i = blockIdx.x * 256 + threadIdx.x; i < n; i += gridDim.x * 256)
    out[i] = (_Float16)in[i];
}

// com_w (216,256,3,3): [m][c*9+tap] -> [m][tap*256+c] so A fragments are contiguous
__global__ __launch_bounds__(256) void cvt_com_kernel(const float* __restrict__ in,
                                                      _Float16* __restrict__ out, int n) {
  for (int i = blockIdx.x * 256 + threadIdx.x; i < n; i += gridDim.x * 256) {
    int m = i / 2304, rem = i % 2304;
    int tap = rem >> 8, c = rem & 255;
    out[i] = (_Float16)in[m * 2304 + c * 9 + tap];
  }
}

// global average pool: one block per (b,c)
__global__ __launch_bounds__(256) void pool_kernel(const float* __restrict__ x,
                                                   float* __restrict__ pooled) {
  __shared__ float s[256];
  const float* p = x + (size_t)blockIdx.x * HW;
  float acc = 0.f;
  for (int i = threadIdx.x; i < HW; i += 256) acc += p[i];
  s[threadIdx.x] = acc; __syncthreads();
  for (int st = 128; st > 0; st >>= 1) {
    if (threadIdx.x < st) s[threadIdx.x] += s[threadIdx.x + st];
    __syncthreads();
  }
  if (threadIdx.x == 0) pooled[blockIdx.x] = s[0] * (1.f / HW);
}

// attention branch: GEMV(256x256) + GN(32 groups of 8) + sigmoid. grid = B.
__global__ __launch_bounds__(256) void atten_kernel(const float* __restrict__ pooled,
                                                    const float* __restrict__ aw,
                                                    const float* __restrict__ ab,
                                                    const float* __restrict__ g1w,
                                                    const float* __restrict__ g1b,
                                                    float* __restrict__ atten) {
  __shared__ float sp[256];
  __shared__ float sv[256];
  int b = blockIdx.x, o = threadIdx.x;
  sp[o] = pooled[b * 256 + o];
  __syncthreads();
  float acc = ab[o];
  for (int k = 0; k < 256; ++k) acc += aw[o * 256 + k] * sp[k];
  sv[o] = acc; __syncthreads();
  int g0 = (o >> 3) << 3;
  float m = 0.f, v = 0.f;
  for (int j = 0; j < 8; ++j) { float t = sv[g0 + j]; m += t; v += t * t; }
  m *= 0.125f; v = v * 0.125f - m * m;
  float xn = (acc - m) * rsqrtf(v + EPSV) * g1w[o] + g1b[o];
  atten[b * 256 + o] = sigm(xn);
}

// x_h[p][c] = f16( feat_l * (1 + atten[b][c]) )   (pixel-major)
__global__ __launch_bounds__(256) void stage_x_kernel(const float* __restrict__ fl,
                                                      const float* __restrict__ atten,
                                                      _Float16* __restrict__ xh, int total) {
  for (int i = blockIdx.x * 256 + threadIdx.x; i < total; i += gridDim.x * 256) {
    int hw = i % HW; int cb = i / HW; int c = cb & 255; int b = cb >> 8;
    xh[((size_t)b * HW + hw) * 256 + c] = (_Float16)(fl[i] * (1.f + atten[b * 256 + c]));
  }
}

// bilinear upsample 48x48 -> 96x96: f32 channel-major (for dcn gathers) and
// f16(2*val) into pixel-major cat[:, 256:512]
__global__ __launch_bounds__(256) void upsample_kernel(const float* __restrict__ fs,
                                                       float* __restrict__ up,
                                                       _Float16* __restrict__ cath, int total) {
  for (int i = blockIdx.x * 256 + threadIdx.x; i < total; i += gridDim.x * 256) {
    int hw = i % HW; int bc = i / HW;
    int h = hw / WID, w = hw % WID;
    float sy = fmaxf((h + 0.5f) * 0.5f - 0.5f, 0.f);
    float sx = fmaxf((w + 0.5f) * 0.5f - 0.5f, 0.f);
    int y0 = (int)sy, x0 = (int)sx;
    int y1 = imin(y0 + 1, 47), x1 = imin(x0 + 1, 47);
    float fy = sy - (float)y0, fx = sx - (float)x0;
    const float* p = fs + (size_t)bc * 48 * 48;
    float v = p[y0 * 48 + x0] * (1.f - fy) * (1.f - fx)
            + p[y0 * 48 + x1] * (1.f - fy) * fx
            + p[y1 * 48 + x0] * fy * (1.f - fx)
            + p[y1 * 48 + x1] * fy * fx;
    up[i] = v;
    int b = bc >> 8, c = bc & 255;
    cath[((size_t)b * HW + hw) * 512 + 256 + c] = (_Float16)(2.f * v);
  }
}

// GroupNorm stats: grid = B*32; each group spans 8 channels * HW (channel-major f32)
__global__ __launch_bounds__(256) void gn_stats_kernel(const float* __restrict__ x,
                                                       float* __restrict__ stats) {
  __shared__ float ss[256], s2[256];
  int bg = blockIdx.x; int b = bg >> 5; int g = bg & 31;
  const float* p = x + ((size_t)b * 256 + g * 8) * HW;
  const int n = 8 * HW;
  float a = 0.f, q = 0.f;
  for (int i = threadIdx.x; i < n; i += 256) { float t = p[i]; a += t; q += t * t; }
  ss[threadIdx.x] = a; s2[threadIdx.x] = q; __syncthreads();
  for (int st = 128; st > 0; st >>= 1) {
    if (threadIdx.x < st) { ss[threadIdx.x] += ss[threadIdx.x + st]; s2[threadIdx.x] += s2[threadIdx.x + st]; }
    __syncthreads();
  }
  if (threadIdx.x == 0) {
    float m = ss[0] / n;
    float v = s2[0] / n - m * m;
    stats[bg * 2]     = m;
    stats[bg * 2 + 1] = rsqrtf(v + EPSV);
  }
}

// GN apply; optional f32 out (channel-major, in-place ok) and/or f16 out
// (pixel-major, row stride ld16, channel offset coff)
__global__ __launch_bounds__(256) void gn_apply_kernel(const float* __restrict__ x,
                                                       const float* __restrict__ stats,
                                                       const float* __restrict__ gw,
                                                       const float* __restrict__ gb,
                                                       float* __restrict__ of32,
                                                       _Float16* __restrict__ of16,
                                                       int ld16, int coff, int total) {
  for (int i = blockIdx.x * 256 + threadIdx.x; i < total; i += gridDim.x * 256) {
    int hw = i % HW; int cb = i / HW; int c = cb & 255; int b = cb >> 8;
    int bg = b * 32 + (c >> 3);
    float xn = (x[i] - stats[bg * 2]) * stats[bg * 2 + 1] * gw[c] + gb[c];
    if (of32) of32[i] = xn;
    if (of16) of16[((size_t)b * HW + hw) * ld16 + coff + c] = (_Float16)xn;
  }
}

// ----------------------------- WMMA GEMM kernels ----------------------------

// 1x1 conv GEMM: out[b][m][p] = sum_k Wh[m][k] * Xp[b*HW+p][k] + bias[m]
// 4 N-subtiles per wave (A fragment reused 4x per K step).
#define GNT 4
__global__ __launch_bounds__(32) void gemm1x1_kernel(const _Float16* __restrict__ Wh,
                                                     const _Float16* __restrict__ Xp,
                                                     const float* __restrict__ bias,
                                                     float* __restrict__ out,
                                                     int M, int K) {
  const int lane = threadIdx.x, half = lane >> 4, l15 = lane & 15;
  const int tpb = HW / (16 * GNT);              // 144 tiles per batch
  const int b = blockIdx.x / tpb;
  const int p0 = (blockIdx.x % tpb) * (16 * GNT);
  const int mtile = blockIdx.y;
  const _Float16* Arow  = Wh + (size_t)(mtile * 16 + l15) * K;
  const _Float16* Brow0 = Xp + ((size_t)b * HW + p0 + l15) * K;
  v8f acc[GNT] = {};
  for (int k0 = 0; k0 < K; k0 += 32) {
    if (k0 + 64 < K) __builtin_prefetch(Arow + k0 + 64, 0, 1);
    v16h a = load_a_frag(Arow + k0, half);
#pragma unroll
    for (int s = 0; s < GNT; ++s) {
      v16h bf = load_b_frag(Brow0 + (size_t)(s * 16) * K + k0, half);
      acc[s] = __builtin_amdgcn_wmma_f32_16x16x32_f16(false, a, false, bf, (short)0, acc[s], false, false);
    }
  }
#pragma unroll
  for (int s = 0; s < GNT; ++s)
#pragma unroll
    for (int r = 0; r < 8; ++r) {
      int m = mtile * 16 + half * 8 + r;
      out[((size_t)b * M + m) * HW + p0 + s * 16 + l15] = acc[s][r] + bias[m];
    }
}

// 3x3 conv: one full image row (6 subtiles of 16 pixels) per wave; 9 shifted
// K=256 GEMMs with uniform tap skip. Wt: [m][tap*256+c]. Xp: [B*HW][256].
__global__ __launch_bounds__(32) void conv3x3_kernel(const _Float16* __restrict__ Wt,
                                                     const _Float16* __restrict__ Xp,
                                                     const float* __restrict__ bias,
                                                     float* __restrict__ out) {
  const int Co = 216;
  const int lane = threadIdx.x, half = lane >> 4, l15 = lane & 15;
  const int row = blockIdx.x;                   // 0..B*96-1
  const int b = row / WID, h0 = row % WID;
  const int mtile = blockIdx.y;
  const int m_a = imin(mtile * 16 + l15, Co - 1);   // clamp: rows >=216 unstored
  const _Float16* Arow = Wt + (size_t)m_a * 2304;
  const _Float16* Xrow = Xp + ((size_t)b * HW + h0 * WID) * 256;
  v8f acc[6] = {};
  for (int tap = 0; tap < 9; ++tap) {
    int dh = tap / 3 - 1, dw = tap % 3 - 1;
    int hh = h0 + dh;
    if (hh < 0 || hh >= WID) continue;          // uniform skip (whole row)
    const _Float16* Srow = Xrow + (size_t)(dh * WID + dw) * 256;
    for (int k0 = 0; k0 < 256; k0 += 32) {
      v16h a = load_a_frag(Arow + tap * 256 + k0, half);
#pragma unroll
      for (int s = 0; s < 6; ++s) {
        int wn = s * 16 + l15 + dw;
        v16h bf;
        if (wn >= 0 && wn < WID)
          bf = load_b_frag(Srow + (size_t)(s * 16 + l15) * 256 + k0, half);
        else
          bf = zero16();
        acc[s] = __builtin_amdgcn_wmma_f32_16x16x32_f16(false, a, false, bf, (short)0, acc[s], false, false);
      }
    }
  }
#pragma unroll
  for (int s = 0; s < 6; ++s)
#pragma unroll
    for (int r = 0; r < 8; ++r) {
      int m = mtile * 16 + half * 8 + r;
      if (m < Co) out[((size_t)b * Co + m) * HW + h0 * WID + s * 16 + l15] = acc[s][r] + bias[m];
    }
}

// Modulated deformable conv as GEMM M=256, K=2304 (k = cidx*9+tap), N=16 pixels.
// Bilinear weights*mask and clipped indices precomputed per (g,tap,pixel) in LDS
// (shared across the 32 channels of each deform group). Epilogue: relu + arm.
__global__ __launch_bounds__(32) void dcn_kernel(const _Float16* __restrict__ Wh,
                                                 const float* __restrict__ up,
                                                 const float* __restrict__ om,
                                                 const float* __restrict__ bias,
                                                 const float* __restrict__ arm,
                                                 float* __restrict__ out) {
  __shared__ float swgt[8 * 9 * 16 * 4];
  __shared__ int   sidx[8 * 9 * 16 * 4];
  const int lane = threadIdx.x, half = lane >> 4, l15 = lane & 15;
  const int ntile = blockIdx.x, mtile = blockIdx.y;
  const int b = ntile / (HW / 16);
  const int p0 = (ntile % (HW / 16)) * 16;
  const int h0 = p0 / WID, w0 = p0 % WID;
  const float* omb = om + (size_t)b * 216 * HW;

  for (int id = lane; id < 8 * 9 * 16; id += 32) {
    int g = id / 144, rem = id % 144, k = rem / 16, n = rem % 16;
    int p = p0 + n;
    float oy = omb[(size_t)(g * 18 + 2 * k)     * HW + p];
    float ox = omb[(size_t)(g * 18 + 2 * k + 1) * HW + p];
    float mk = sigm(omb[(size_t)(144 + g * 9 + k) * HW + p]);
    int kh = k / 3, kw = k % 3;
    float py = oy + (float)(h0 - 1 + kh);
    float px = ox + (float)(w0 + n - 1 + kw);
    float y0f = floorf(py), x0f = floorf(px);
    float fy = py - y0f, fx = px - x0f;
#pragma unroll
    for (int corner = 0; corner < 4; ++corner) {
      int dy = corner >> 1, dx = corner & 1;
      float yi = y0f + (float)dy, xi = x0f + (float)dx;
      float wv = (dy ? fy : 1.f - fy) * (dx ? fx : 1.f - fx);
      bool ok = (yi >= 0.f) && (yi <= 95.f) && (xi >= 0.f) && (xi <= 95.f);
      int iy = imin(imax((int)yi, 0), 95);
      int ix = imin(imax((int)xi, 0), 95);
      swgt[id * 4 + corner] = ok ? wv * mk : 0.f;
      sidx[id * 4 + corner] = iy * WID + ix;
    }
  }
  __syncthreads();

  const float* upb = up + (size_t)b * 256 * HW;
  const _Float16* Arow = Wh + (size_t)(mtile * 16 + l15) * 2304;
  v8f c = {};
  for (int kk = 0; kk < 2304; kk += 32) {
    if (kk + 64 < 2304) __builtin_prefetch(Arow + kk + 64, 0, 1);
    v16h a = load_a_frag(Arow + kk, half);
    v16h bf;
#pragma unroll
    for (int e = 0; e < 16; ++e) {
      int kf   = kk + half * 16 + e;        // B elem e holds K = half*16 + e
      int cidx = kf / 9;
      int tap  = kf - cidx * 9;
      int g    = cidx >> 5;
      int ent  = (((g * 9 + tap) * 16) + l15) * 4;
      const float* src = upb + (size_t)cidx * HW;
      float s = swgt[ent]     * src[sidx[ent]]
              + swgt[ent + 1] * src[sidx[ent + 1]]
              + swgt[ent + 2] * src[sidx[ent + 2]]
              + swgt[ent + 3] * src[sidx[ent + 3]];
      bf[e] = (_Float16)s;
    }
    c = __builtin_amdgcn_wmma_f32_16x16x32_f16(false, a, false, bf, (short)0, c, false, false);
  }
#pragma unroll
  for (int r = 0; r < 8; ++r) {
    int m = mtile * 16 + half * 8 + r;
    size_t o = ((size_t)b * 256 + m) * HW + p0 + l15;
    out[o] = fmaxf(c[r] + bias[m], 0.f) + arm[o];
  }
}

// --------------------------------- launcher ---------------------------------

extern "C" void kernel_launch(void* const* d_in, const int* in_sizes, int n_in,
                              void* d_out, int out_size, void* d_ws, size_t ws_size,
                              hipStream_t stream) {
  (void)in_sizes; (void)n_in; (void)out_size; (void)ws_size;
  const float* feat_l      = (const float*)d_in[0];
  const float* feat_s      = (const float*)d_in[1];
  const float* fsm_atten_w = (const float*)d_in[2];
  const float* fsm_atten_b = (const float*)d_in[3];
  const float* fsm_gn1_w   = (const float*)d_in[4];
  const float* fsm_gn1_b   = (const float*)d_in[5];
  const float* fsm_conv_w  = (const float*)d_in[6];
  const float* fsm_conv_b  = (const float*)d_in[7];
  const float* fsm_gn2_w   = (const float*)d_in[8];
  const float* fsm_gn2_b   = (const float*)d_in[9];
  const float* off_w       = (const float*)d_in[10];
  const float* off_b       = (const float*)d_in[11];
  const float* gn_off_w    = (const float*)d_in[12];
  const float* gn_off_b    = (const float*)d_in[13];
  const float* com_w       = (const float*)d_in[14];
  const float* com_b       = (const float*)d_in[15];
  const float* dcn_w       = (const float*)d_in[16];
  const float* dcn_b       = (const float*)d_in[17];
  float* outp = (float*)d_out;

  // workspace layout (bytes); ~88 MB total
  const size_t BCHW = (size_t)NB * NC * HW;       // 4,718,592 elems
  char* ws = (char*)d_ws;
  size_t o = 0;
  float*    FEAT_UP = (float*)(ws + o);           o += BCHW * 4;   // f32 [b][c][hw]
  _Float16* XH      = (_Float16*)(ws + o);        o += BCHW * 2;   // f16 [b*HW+p][256]
  float*    ARM     = (float*)(ws + o);           o += BCHW * 4;   // f32 [b][c][hw]
  _Float16* CATH    = (_Float16*)(ws + o);        o += 2 * BCHW * 2; // f16 [b*HW+p][512]
  float*    OFFRAW  = (float*)(ws + o);           o += BCHW * 4;   // reused as OM
  float*    POOLED  = (float*)(ws + o);           o += 512 * 4;
  float*    ATTEN   = (float*)(ws + o);           o += 512 * 4;
  float*    STATS1  = (float*)(ws + o);           o += 64 * 2 * 4;
  float*    STATS2  = (float*)(ws + o);           o += 64 * 2 * 4;
  _Float16* WFSM_H  = (_Float16*)(ws + o);        o += 65536 * 2;
  _Float16* WOFF_H  = (_Float16*)(ws + o);        o += 131072 * 2;
  _Float16* WCOM_H  = (_Float16*)(ws + o);        o += 497664 * 2;
  _Float16* WDCN_H  = (_Float16*)(ws + o);        o += 589824 * 2;
  _Float16* OFFN_H  = CATH;     // alias: cat dead after off GEMM  [b*HW+p][256]
  float*    OM      = OFFRAW;   // alias: off_raw dead after its GN apply

  const int EW = 256;

  // weights -> f16 (com_w also transposed to [m][tap*256+c])
  cvt_f16_kernel<<<256, EW, 0, stream>>>(fsm_conv_w, WFSM_H, 65536);
  cvt_f16_kernel<<<512, EW, 0, stream>>>(off_w,      WOFF_H, 131072);
  cvt_com_kernel<<<1024, EW, 0, stream>>>(com_w,     WCOM_H, 497664);
  cvt_f16_kernel<<<1024, EW, 0, stream>>>(dcn_w,     WDCN_H, 589824);

  // squeeze-excite branch
  pool_kernel<<<NB * NC, EW, 0, stream>>>(feat_l, POOLED);
  atten_kernel<<<NB, EW, 0, stream>>>(POOLED, fsm_atten_w, fsm_atten_b,
                                      fsm_gn1_w, fsm_gn1_b, ATTEN);

  // stage activations (pixel-major f16)
  stage_x_kernel<<<4096, EW, 0, stream>>>(feat_l, ATTEN, XH, (int)BCHW);
  upsample_kernel<<<4096, EW, 0, stream>>>(feat_s, FEAT_UP, CATH, (int)BCHW);

  // fsm 1x1 conv -> GN -> feat_arm (f32, in-place) + cat[:, 0:256] (f16)
  {
    dim3 g((NB * HW) / (16 * GNT), NC / 16);
    gemm1x1_kernel<<<g, 32, 0, stream>>>(WFSM_H, XH, fsm_conv_b, ARM, NC, 256);
  }
  gn_stats_kernel<<<NB * 32, EW, 0, stream>>>(ARM, STATS1);
  gn_apply_kernel<<<4096, EW, 0, stream>>>(ARM, STATS1, fsm_gn2_w, fsm_gn2_b,
                                           ARM, CATH, 512, 0, (int)BCHW);

  // off 1x1 conv (K=512) -> GN -> offn (f16 pixel-major, overwrites cat)
  {
    dim3 g((NB * HW) / (16 * GNT), NC / 16);
    gemm1x1_kernel<<<g, 32, 0, stream>>>(WOFF_H, CATH, off_b, OFFRAW, NC, 512);
  }
  gn_stats_kernel<<<NB * 32, EW, 0, stream>>>(OFFRAW, STATS2);
  gn_apply_kernel<<<4096, EW, 0, stream>>>(OFFRAW, STATS2, gn_off_w, gn_off_b,
                                           nullptr, OFFN_H, 256, 0, (int)BCHW);

  // 3x3 conv to offsets/masks (216 channels), overwrites OFFRAW region as OM
  {
    dim3 g(NB * WID, 14);   // one image row per block; ceil(216/16) mtiles
    conv3x3_kernel<<<g, 32, 0, stream>>>(WCOM_H, OFFN_H, com_b, OM);
  }

  // modulated deformable conv + relu + residual
  {
    dim3 g((NB * HW) / 16, NC / 16);
    dcn_kernel<<<g, 32, 0, stream>>>(WDCN_H, FEAT_UP, OM, dcn_b, ARM, outp);
  }
}
